// MultiHeadAttention_83124797046813
// MI455X (gfx1250) — matrix operand order repaired and checked by reading, not compiled, not run on hardware
//
#include <hip/hip_runtime.h>
#include <hip/hip_bf16.h>

// ---------------------------------------------------------------------------
// MultiHeadAttention for MI455X (gfx1250): all matmuls via v_wmma_f32_16x16x32_f16.
// Round 2: software-pipelined fragment prefetch (avoid s_wait_loadcnt 0 before
// every WMMA) and 2x register blocking (32 rows/wave) for 2x WMMA-per-load and
// half the L2 weight/K/V traffic.
// ---------------------------------------------------------------------------

typedef _Float16 half_t;
typedef __attribute__((ext_vector_type(16))) _Float16 v16h;
typedef __attribute__((ext_vector_type(8)))  _Float16 v8h;
typedef __attribute__((ext_vector_type(4)))  _Float16 v4h;
typedef __attribute__((ext_vector_type(8)))  float    v8f;
typedef __attribute__((ext_vector_type(4)))  float    v4f;
typedef __attribute__((ext_vector_type(4)))  int      v4i;

constexpr int BB = 4;
constexpr int LQ = 2048;
constexpr int LK = 2048;
constexpr int DM = 1024;
constexpr int NH = 16;
constexpr int DK = 64;
constexpr int MROWS = BB * LQ; // 8192

__device__ __forceinline__ v8f wmma_f16(v16h a, v16h b, v8f c) {
  return __builtin_amdgcn_wmma_f32_16x16x32_f16(false, a, false, b, (short)0, c,
                                                false, false);
}

__device__ __forceinline__ v16h cat16(v8h a, v8h b) {
  return __builtin_shufflevector(a, b, 0, 1, 2, 3, 4, 5, 6, 7, 8, 9, 10, 11, 12,
                                 13, 14, 15);
}

// A-fragment (16x32 f16): M on lanes; halves 0-15 = K(16*hi .. +15), contiguous.
__device__ __forceinline__ v16h load_fragA(const half_t* __restrict__ row, int hi) {
  v8h a = *(const v8h*)(row + 16 * hi);
  v8h b = *(const v8h*)(row + 16 * hi + 8);
  return cat16(a, b);
}
// B-fragment (32x16 f16): N on lanes; halves 0-15 = K(16*hi .. +15), contiguous.
__device__ __forceinline__ v16h load_fragB(const half_t* __restrict__ row, int hi) {
  v8h a = *(const v8h*)(row + 16 * hi);
  v8h b = *(const v8h*)(row + 16 * hi + 8);
  return cat16(a, b);
}

__device__ __forceinline__ float fast_exp2(float x) {
#if __has_builtin(__builtin_amdgcn_exp2f)
  return __builtin_amdgcn_exp2f(x);
#else
  return exp2f(x);
#endif
}
__device__ __forceinline__ float fast_rcp(float x) {
#if __has_builtin(__builtin_amdgcn_rcpf)
  return __builtin_amdgcn_rcpf(x);
#else
  return 1.0f / x;
#endif
}

// --------------------------- f32 -> f16 convert ----------------------------
__global__ __launch_bounds__(256) void cvt_f32_to_f16(const float* __restrict__ src,
                                                      half_t* __restrict__ dst) {
  size_t i = (size_t)blockIdx.x * blockDim.x + threadIdx.x;
  v4f v = *(const v4f*)(src + 4 * i);
  v4h h;
  h[0] = (half_t)v[0]; h[1] = (half_t)v[1];
  h[2] = (half_t)v[2]; h[3] = (half_t)v[3];
  *(v4h*)(dst + 4 * i) = h;
}

// ------------------ weight transpose + f16 (LDS tiled) ---------------------
__global__ __launch_bounds__(256) void transpose_w_f16(const float* __restrict__ w,
                                                       half_t* __restrict__ wT) {
  __shared__ float tile[32][33];
  const int bx = blockIdx.x * 32; // n
  const int by = blockIdx.y * 32; // k
  const int tx = threadIdx.x, ty = threadIdx.y; // 32 x 8
#pragma unroll
  for (int i = 0; i < 32; i += 8)
    tile[ty + i][tx] = w[(size_t)(by + ty + i) * DM + bx + tx];
  __syncthreads();
#pragma unroll
  for (int i = 0; i < 32; i += 8)
    wT[(size_t)(bx + ty + i) * DM + by + tx] = (half_t)tile[tx][ty + i];
}

// --------------------------- WMMA GEMM, C^T path ---------------------------
// Wave computes 32 m-rows x 128 n-cols of D = (A @ W)^T via wmma(W^T-row, A-row).
// Prefetch chain: next W fragment is loaded before the current one is consumed.
template <int STORE_F32>
__global__ __launch_bounds__(256) void gemm_ct_kernel(
    const half_t* __restrict__ A, const half_t* __restrict__ WT,
    half_t* __restrict__ out16, float* __restrict__ outf,
    const float* __restrict__ resid) {
  const int wid  = blockIdx.x * 8 + (threadIdx.x >> 5);
  const int lane = threadIdx.x & 31;
  const int li = lane & 15, hi = lane >> 4;
  const int n0 = (wid & 7) * 128;
  const int m0 = (wid >> 3) * 32;
  const half_t* arow0 = A + (size_t)(m0 + li) * DM;
  const half_t* arow1 = arow0 + (size_t)16 * DM;
  const half_t* wbase = WT + (size_t)(n0 + li) * DM;

  v8f acc[2][8] = {};
  v16h bf0 = load_fragB(arow0, hi);
  v16h bf1 = load_fragB(arow1, hi);
  v16h af  = load_fragA(wbase, hi);

  for (int k = 0; k < DM; k += 32) {
    const bool lastk = (k + 32 >= DM);
    v16h bf0n = bf0, bf1n = bf1;
    if (!lastk) {
      bf0n = load_fragB(arow0 + k + 32, hi);
      bf1n = load_fragB(arow1 + k + 32, hi);
    }
#pragma unroll
    for (int t = 0; t < 8; ++t) {
      v16h afn = af;
      if (t < 7)
        afn = load_fragA(wbase + (size_t)(t + 1) * 16 * DM + k, hi);
      else if (!lastk)
        afn = load_fragA(wbase + k + 32, hi);
      acc[0][t] = wmma_f16(af, bf0, acc[0][t]);
      acc[1][t] = wmma_f16(af, bf1, acc[1][t]);
      af = afn;
    }
    bf0 = bf0n;
    bf1 = bf1n;
  }

  // D: lane column m = m0 + mt*16 + li ; register rows n = n0 + t*16 + 8*hi + j
  if (STORE_F32) {
#pragma unroll
    for (int mt = 0; mt < 2; ++mt) {
      float*       orow = outf  + (size_t)(m0 + mt * 16 + li) * DM + n0 + 8 * hi;
      const float* rrow = resid + (size_t)(m0 + mt * 16 + li) * DM + n0 + 8 * hi;
#pragma unroll
      for (int t = 0; t < 8; ++t) {
        v4f ra = *(const v4f*)(rrow + t * 16);
        v4f rb = *(const v4f*)(rrow + t * 16 + 4);
        v4f oa, ob;
#pragma unroll
        for (int j = 0; j < 4; ++j) {
          oa[j] = acc[mt][t][j] + ra[j];
          ob[j] = acc[mt][t][j + 4] + rb[j];
        }
        *(v4f*)(orow + t * 16)     = oa;
        *(v4f*)(orow + t * 16 + 4) = ob;
      }
    }
  } else {
#pragma unroll
    for (int mt = 0; mt < 2; ++mt) {
      half_t* orow = out16 + (size_t)(m0 + mt * 16 + li) * DM + n0 + 8 * hi;
#pragma unroll
      for (int t = 0; t < 8; ++t) {
        v8h r;
#pragma unroll
        for (int j = 0; j < 8; ++j) r[j] = (half_t)acc[mt][t][j];
        *(v8h*)(orow + t * 16) = r;
      }
    }
  }
}

// ------------------- WMMA GEMM, C path -> transposed V ---------------------
// Vt[b, h, d, lk] = (v @ w_vs)[b*LK+lk, h*64+d]; lane carries n so the store
// lands contiguously in Vt.  32 m-rows per wave, prefetch chain on W fragments.
__global__ __launch_bounds__(256) void gemm_vt_kernel(const half_t* __restrict__ A,
                                                      const half_t* __restrict__ WT,
                                                      half_t* __restrict__ Vt) {
  const int wid  = blockIdx.x * 8 + (threadIdx.x >> 5);
  const int lane = threadIdx.x & 31;
  const int li = lane & 15, hi = lane >> 4;
  const int n0 = (wid & 7) * 128;
  const int m0 = (wid >> 3) * 32;
  const half_t* arow0 = A + (size_t)(m0 + li) * DM;
  const half_t* arow1 = arow0 + (size_t)16 * DM;
  const half_t* wbase = WT + (size_t)(n0 + li) * DM;

  v8f acc[2][8] = {};
  v16h af0 = load_fragA(arow0, hi);
  v16h af1 = load_fragA(arow1, hi);
  v16h bf  = load_fragB(wbase, hi);

  for (int k = 0; k < DM; k += 32) {
    const bool lastk = (k + 32 >= DM);
    v16h af0n = af0, af1n = af1;
    if (!lastk) {
      af0n = load_fragA(arow0 + k + 32, hi);
      af1n = load_fragA(arow1 + k + 32, hi);
    }
#pragma unroll
    for (int t = 0; t < 8; ++t) {
      v16h bfn = bf;
      if (t < 7)
        bfn = load_fragB(wbase + (size_t)(t + 1) * 16 * DM + k, hi);
      else if (!lastk)
        bfn = load_fragB(wbase + k + 32, hi);
      acc[0][t] = wmma_f16(af0, bf, acc[0][t]);
      acc[1][t] = wmma_f16(af1, bf, acc[1][t]);
      bf = bfn;
    }
    af0 = af0n;
    af1 = af1n;
  }

  const int bb = m0 >> 11; // batch index (LK rows per batch)
#pragma unroll
  for (int mt = 0; mt < 2; ++mt) {
    const int lk = (m0 & (LK - 1)) + mt * 16 + 8 * hi;
#pragma unroll
    for (int t = 0; t < 8; ++t) {
      const int n = n0 + t * 16 + li; // = h*64 + d
      v8h r;
#pragma unroll
      for (int j = 0; j < 8; ++j) r[j] = (half_t)acc[mt][t][j];
      *(v8h*)(Vt + ((size_t)bb * DM + n) * LK + lk) = r;
    }
  }
}

// ------------------------- flash attention (1 wave / 32 q-rows) ------------
// S^T = K_tile x Q^T (q on lanes, keys on regs) -> in-lane online softmax with
// a single shfl_xor(16) per reduction; O^T = V^T x P (q stays on lanes).
// K/V fragments are shared across the wave's two q-tiles; mask is loaded per q-tile.
__global__ __launch_bounds__(256) void attn_kernel(const half_t* __restrict__ Qh,
                                                   const half_t* __restrict__ Kh,
                                                   const half_t* __restrict__ Vt,
                                                   const int* __restrict__ mask,
                                                   half_t* __restrict__ O) {
  const int wid  = blockIdx.x * 8 + (threadIdx.x >> 5);
  const int lane = threadIdx.x & 31;
  const int li = lane & 15, hi = lane >> 4;
  const int qt = wid & (LQ / 32 - 1);
  const int h  = (wid >> 6) & (NH - 1);
  const int b  = wid >> 10;
  constexpr float SCALE = 0.125f * 1.44269504088896f; // 1/sqrt(64) * log2(e)

  v16h qf[2][2];
  const int* mrow[2];
#pragma unroll
  for (int u = 0; u < 2; ++u) {
    const int q = qt * 32 + u * 16 + li;
    const half_t* qrow = Qh + ((size_t)b * LQ + q) * DM + h * DK;
    qf[u][0] = load_fragB(qrow, hi);
    qf[u][1] = load_fragB(qrow + 32, hi);
    mrow[u] = mask + ((size_t)b * LQ + q) * LK;
  }
  const half_t* vbase = Vt + (((size_t)b * NH + h) * DK + li) * LK;

  v8f acc[2][4] = {};
  float row_m[2] = {-1e30f, -1e30f};
  float row_l[2] = {0.0f, 0.0f};

  for (int kb = 0; kb < LK; kb += 32) {
    const half_t* krow0 = Kh + ((size_t)b * LK + kb + li) * DM + h * DK;
    const half_t* krow1 = krow0 + (size_t)16 * DM;
    const v16h kf00 = load_fragA(krow0, hi);
    const v16h kf01 = load_fragA(krow0 + 32, hi);
    const v16h kf10 = load_fragA(krow1, hi);
    const v16h kf11 = load_fragA(krow1 + 32, hi);

    // V fragments issued early so they overlap the softmax VALU work.
    const half_t* vrow = vbase + kb;
    const v16h vf0 = load_fragA(vrow, hi);
    const v16h vf1 = load_fragA(vrow + (size_t)16 * LK, hi);
    const v16h vf2 = load_fragA(vrow + (size_t)32 * LK, hi);
    const v16h vf3 = load_fragA(vrow + (size_t)48 * LK, hi);

    v16h pf[2];
#pragma unroll
    for (int u = 0; u < 2; ++u) {
      v8f s0 = {}, s1 = {};
      s0 = wmma_f16(kf00, qf[u][0], s0);
      s0 = wmma_f16(kf01, qf[u][1], s0);
      s1 = wmma_f16(kf10, qf[u][0], s1);
      s1 = wmma_f16(kf11, qf[u][1], s1);

      // lane's 16 mask values: keys kb + 8*hi + {0..7, 16..23}
      const int* mp = mrow[u] + kb + 8 * hi;
      v4i ma = *(const v4i*)(mp);
      v4i mb = *(const v4i*)(mp + 4);
      v4i mc = *(const v4i*)(mp + 16);
      v4i md = *(const v4i*)(mp + 20);

      float r = -1e30f;
#pragma unroll
      for (int j = 0; j < 8; ++j) {
        s0[j] *= SCALE;
        s1[j] *= SCALE;
        r = fmaxf(r, fmaxf(s0[j], s1[j]));
      }
      r = fmaxf(r, __shfl_xor(r, 16, 32));
      const float mnew  = fmaxf(row_m[u], r);
      const float alpha = fast_exp2(row_m[u] - mnew);
      row_m[u] = mnew;

      union HU { v8h h8; unsigned uu[4]; };
      HU p0, p1, o0, o1;
      float t = 0.0f;
#pragma unroll
      for (int j = 0; j < 8; ++j) {
        const int mv0 = (j < 4) ? ma[j] : mb[j - 4];
        const int mv1 = (j < 4) ? mc[j] : md[j - 4];
        const float e0 = mv0 ? fast_exp2(s0[j] - mnew) : 0.0f;
        const float e1 = mv1 ? fast_exp2(s1[j] - mnew) : 0.0f;
        t += e0 + e1;
        p0.h8[j] = (half_t)e0;
        p1.h8[j] = (half_t)e1;
      }
      t += __shfl_xor(t, 16, 32);
      row_l[u] = row_l[u] * alpha + t;
#pragma unroll
      for (int j = 0; j < 8; ++j) {
        acc[u][0][j] *= alpha; acc[u][1][j] *= alpha;
        acc[u][2][j] *= alpha; acc[u][3][j] *= alpha;
      }
      // reshape P into B-fragment key order (lane<16: keys 0..15; lane>=16: 16..31)
#pragma unroll
      for (int i = 0; i < 4; ++i) {
        o0.uu[i] = __shfl_xor(p0.uu[i], 16, 32);
        o1.uu[i] = __shfl_xor(p1.uu[i], 16, 32);
      }
      const v8h plo = hi ? o1.h8 : p0.h8;
      const v8h phi = hi ? p1.h8 : o0.h8;
      pf[u] = cat16(plo, phi);
    }

#pragma unroll
    for (int u = 0; u < 2; ++u) {
      acc[u][0] = wmma_f16(vf0, pf[u], acc[u][0]);
      acc[u][1] = wmma_f16(vf1, pf[u], acc[u][1]);
      acc[u][2] = wmma_f16(vf2, pf[u], acc[u][2]);
      acc[u][3] = wmma_f16(vf3, pf[u], acc[u][3]);
    }
  }

#pragma unroll
  for (int u = 0; u < 2; ++u) {
    const int q = qt * 32 + u * 16 + li;
    const float inv = fast_rcp(row_l[u]);
    half_t* orow = O + ((size_t)b * LQ + q) * DM + h * DK + 8 * hi;
    v8h r0, r1, r2, r3;
#pragma unroll
    for (int j = 0; j < 8; ++j) {
      r0[j] = (half_t)(acc[u][0][j] * inv);
      r1[j] = (half_t)(acc[u][1][j] * inv);
      r2[j] = (half_t)(acc[u][2][j] * inv);
      r3[j] = (half_t)(acc[u][3][j] * inv);
    }
    *(v8h*)(orow)      = r0;
    *(v8h*)(orow + 16) = r1;
    *(v8h*)(orow + 32) = r2;
    *(v8h*)(orow + 48) = r3;
  }
}

// ------------------------------- launcher ----------------------------------
extern "C" void kernel_launch(void* const* d_in, const int* in_sizes, int n_in,
                              void* d_out, int out_size, void* d_ws, size_t ws_size,
                              hipStream_t stream) {
  const float* q    = (const float*)d_in[0];
  const float* k    = (const float*)d_in[1];
  const float* v    = (const float*)d_in[2];
  const int*   mask = (const int*)d_in[3];
  const float* wqs  = (const float*)d_in[4];
  const float* wks  = (const float*)d_in[5];
  const float* wvs  = (const float*)d_in[6];
  const float* wfc  = (const float*)d_in[7];
  float* out = (float*)d_out;

  char* ws = (char*)d_ws;
  const size_t SZ_ACT = (size_t)MROWS * DM * sizeof(half_t); // 16 MB
  const size_t SZ_W   = (size_t)DM * DM * sizeof(half_t);    //  2 MB
  half_t* q16 = (half_t*)(ws);              // later reused as attention output O
  half_t* k16 = (half_t*)(ws + 1 * SZ_ACT);
  half_t* v16 = (half_t*)(ws + 2 * SZ_ACT);
  half_t* Qh  = (half_t*)(ws + 3 * SZ_ACT);
  half_t* Kh  = (half_t*)(ws + 4 * SZ_ACT);
  half_t* Vt  = (half_t*)(ws + 5 * SZ_ACT);
  half_t* wqT = (half_t*)(ws + 6 * SZ_ACT);
  half_t* wkT = (half_t*)(ws + 6 * SZ_ACT + 1 * SZ_W);
  half_t* wvT = (half_t*)(ws + 6 * SZ_ACT + 2 * SZ_W);
  half_t* wfT = (half_t*)(ws + 6 * SZ_ACT + 3 * SZ_W);
  half_t* O   = q16; // q16 is dead after the Q projection

  const int cvtBlocks = (MROWS * DM / 4) / 256; // 8192
  cvt_f32_to_f16<<<cvtBlocks, 256, 0, stream>>>(q, q16);
  cvt_f32_to_f16<<<cvtBlocks, 256, 0, stream>>>(k, k16);
  cvt_f32_to_f16<<<cvtBlocks, 256, 0, stream>>>(v, v16);

  dim3 tb(32, 8), tg(32, 32);
  transpose_w_f16<<<tg, tb, 0, stream>>>(wqs, wqT);
  transpose_w_f16<<<tg, tb, 0, stream>>>(wks, wkT);
  transpose_w_f16<<<tg, tb, 0, stream>>>(wvs, wvT);
  transpose_w_f16<<<tg, tb, 0, stream>>>(wfc, wfT);

  const int gemmBlocks = (MROWS / 32) * (DM / 128) / 8; // 256 blocks x 8 waves
  gemm_ct_kernel<0><<<gemmBlocks, 256, 0, stream>>>(q16, wqT, Qh, nullptr, nullptr);
  gemm_ct_kernel<0><<<gemmBlocks, 256, 0, stream>>>(k16, wkT, Kh, nullptr, nullptr);
  gemm_vt_kernel<<<gemmBlocks, 256, 0, stream>>>(v16, wvT, Vt);

  const int attnBlocks = BB * NH * (LQ / 32) / 8; // 512 blocks x 8 waves
  attn_kernel<<<attnBlocks, 256, 0, stream>>>(Qh, Kh, Vt, mask, O);

  gemm_ct_kernel<1><<<gemmBlocks, 256, 0, stream>>>(O, wfT, nullptr, out, q);
}
